// MoEGCN_11871289606706
// MI455X (gfx1250) — matrix operand order repaired and checked by reading, not compiled, not run on hardware
//
#include <hip/hip_runtime.h>
#include <hip/hip_bf16.h>

// ---------------- problem constants (match reference) ----------------
#define N_NODES 30000
#define N_EDGES 480000
#define IN_F    512
#define HID     256
#define OUT_F   40
#define N_EXP   64
#define CAP     2048
#define BN_EPS  1e-5f

// ---------------- WMMA / vector types ----------------
typedef __bf16 v16bf __attribute__((ext_vector_type(16)));
typedef __bf16 bf8v  __attribute__((ext_vector_type(8)));
typedef __bf16 bf4v  __attribute__((ext_vector_type(4)));
typedef float  v8f   __attribute__((ext_vector_type(8)));

// ---------------- tiled WMMA GEMM ----------------
// C[M,N] = act(A[M,K] @ B[K,N] + bias), batched over blockIdx.z via strides.
// 256 threads = 8 waves; block tile 64x64; each wave owns a 16x32 C strip
// (2 accumulators, 2 wmma per K-step, A fragment reused).
// Software pipeline: tile k+1 staged while WMMA of tile k runs; ping-pong
// LDS; one barrier per K-step; WGP-scope global_prefetch of tile k+2.
// bf16-A fast path stages A via global_load_async_to_lds_b128 (ASYNCcnt DMA);
// f32-A path stages via registers with cvt to bf16.
// K is always a multiple of 32 here -> no K guards.
#define GT   256
#define TMv  64
#define TNv  64
#define TKS  32
#define ASTR (TKS + 8)   // 40 bf16 = 80 B stride (16B-aligned)
#define BSTR (TKS + 8)
#define ABUF (TMv * ASTR)
#define BBUF (TNv * BSTR)

template <typename T> struct stage_t;
template <> struct stage_t<float>  { typedef float4 type; };
template <> struct stage_t<__bf16> { typedef bf4v   type; };

template <typename T> struct is_bf16 { static const bool value = false; };
template <> struct is_bf16<__bf16>  { static const bool value = true; };

template <typename T>
__device__ __forceinline__ T zeroval() { T v = {}; return v; }

__device__ __forceinline__ bf4v to_bf(const float4& v) {
  bf4v r;
  r[0] = (__bf16)v.x; r[1] = (__bf16)v.y; r[2] = (__bf16)v.z; r[3] = (__bf16)v.w;
  return r;
}
__device__ __forceinline__ bf4v to_bf(const bf4v& v) { return v; }

template <bool GUARD, int ACT, typename AT, typename OT>
__device__ __forceinline__ void gemm_body(
    const AT* __restrict__ A, const float* __restrict__ B,
    const float* __restrict__ bp, OT* __restrict__ C,
    int M, int N, int K, __bf16* AsBase, __bf16* BsBase, int bm, int bn)
{
  const int t    = threadIdx.x;
  const int lane = t & 31;
  const int wave = t >> 5;
  const int wr   = wave >> 1;    // 0..3 (row strip of 16)
  const int wc   = wave & 1;     // 0..1 (col strip of 32)
  const int lrow = lane & 15;
  const int half = lane >> 4;

  typedef typename stage_t<AT>::type AS4;
  // bf16 A tiles never hit the guarded path in this workload (M=2048, N%64==0)
  constexpr bool ASYNC_A = (!GUARD) && is_bf16<AT>::value;

  // register-staged A strips: s = {t, t+256} -> row=s>>3, col=(s&7)*4
  const int ar0 = t >> 3,        ac0 = (t & 7) << 2;
  const int ar1 = (t + GT) >> 3, ac1 = ((t + GT) & 7) << 2;
  const int gra0 = bm + ar0, gra1 = bm + ar1;
  // async A strips: 16B per lane, 256 lanes = full 64x32 bf16 tile
  const int aar = t >> 2,        aac = (t & 3) << 3;
  // B strips: 4-deep K columns, coalesced b32
  const int bc0 = t & 63,        bk0 = (t >> 6) << 2;
  const int bc1 = (t + GT) & 63, bk1 = ((t + GT) >> 6) << 2;
  const int gcb0 = bn + bc0, gcb1 = bn + bc1;

  AS4    sa0, sa1;  // staged A (register path)
  float4 sb0, sb1;  // staged B column strips

  auto issueAsyncA = [&](int k0, __bf16* As) {
    if constexpr (ASYNC_A) {
      const AT* gp = &A[(long)(bm + aar) * K + k0 + aac];
      unsigned lds = (unsigned)(size_t)&As[aar * ASTR + aac];
      asm volatile("global_load_async_to_lds_b128 %0, %1, off"
                   :: "v"(lds), "v"(gp) : "memory");
    }
  };
  auto loadTilesReg = [&](int k0) {
    if constexpr (!ASYNC_A) {
      sa0 = zeroval<AS4>(); sa1 = zeroval<AS4>();
      if (!GUARD || gra0 < M) sa0 = *(const AS4*)&A[(long)gra0 * K + k0 + ac0];
      if (!GUARD || gra1 < M) sa1 = *(const AS4*)&A[(long)gra1 * K + k0 + ac1];
    }
    sb0 = zeroval<float4>(); sb1 = zeroval<float4>();
    if (!GUARD || gcb0 < N) {
      const float* p = &B[(long)(k0 + bk0) * N + gcb0];
      sb0.x = p[0]; sb0.y = p[(long)N]; sb0.z = p[2L * N]; sb0.w = p[3L * N];
    }
    if (!GUARD || gcb1 < N) {
      const float* p = &B[(long)(k0 + bk1) * N + gcb1];
      sb1.x = p[0]; sb1.y = p[(long)N]; sb1.z = p[2L * N]; sb1.w = p[3L * N];
    }
    // WGP-scope prefetch of the tile after next (global_prefetch_b8)
    int kpf = k0 + TKS;
    if (kpf < K) {
      if (!GUARD || gra0 < M) __builtin_prefetch(&A[(long)gra0 * K + kpf + ac0], 0, 3);
      if (!GUARD || gcb0 < N) __builtin_prefetch(&B[(long)(kpf + bk0) * N + gcb0], 0, 3);
    }
  };
  auto storeTilesReg = [&](__bf16* As, __bf16* Bs) {
    if constexpr (!ASYNC_A) {
      *(bf4v*)&As[ar0 * ASTR + ac0] = to_bf(sa0);
      *(bf4v*)&As[ar1 * ASTR + ac1] = to_bf(sa1);
    }
    *(bf4v*)&Bs[bc0 * BSTR + bk0] = to_bf(sb0);
    *(bf4v*)&Bs[bc1 * BSTR + bk1] = to_bf(sb1);
  };
  auto waitAsync = [&]() {
    if constexpr (ASYNC_A) {
      asm volatile("s_wait_asynccnt 0x0" ::: "memory");
    }
  };

  v8f acc[2] = {{}, {}};

  issueAsyncA(0, AsBase);
  loadTilesReg(0);
  storeTilesReg(AsBase, BsBase);
  waitAsync();
  __syncthreads();

  int p = 0;
  for (int k0 = 0; k0 < K; k0 += TKS) {
    const bool hasNext = (k0 + TKS) < K;
    if (hasNext) {                      // staging overlaps WMMA below
      issueAsyncA(k0 + TKS, AsBase + (p ^ 1) * ABUF);
      loadTilesReg(k0 + TKS);
    }

    __bf16* As = AsBase + p * ABUF;
    __bf16* Bs = BsBase + p * BBUF;

    // fragments: two b128 LDS reads each (ISA 16-bit A/B layouts)
    const int arow = wr * 16 + lrow;
    bf8v a0 = *(const bf8v*)&As[arow * ASTR + 8 * half];
    bf8v a1 = *(const bf8v*)&As[arow * ASTR + 16 + 8 * half];
    v16bf a = __builtin_shufflevector(a0, a1, 0, 1, 2, 3, 4, 5, 6, 7,
                                      8, 9, 10, 11, 12, 13, 14, 15);
#pragma unroll
    for (int c = 0; c < 2; ++c) {
      const int bcol = wc * 32 + c * 16 + lrow;
      bf8v p0 = *(const bf8v*)&Bs[bcol * BSTR + 8 * half];
      bf8v p1 = *(const bf8v*)&Bs[bcol * BSTR + 16 + 8 * half];
      v16bf b = __builtin_shufflevector(p0, p1, 0, 1, 2, 3, 4, 5, 6, 7,
                                        8, 9, 10, 11, 12, 13, 14, 15);
      acc[c] = __builtin_amdgcn_wmma_f32_16x16x32_bf16(
          false, a, false, b, (short)0, acc[c], false, false);
    }

    if (hasNext) {
      storeTilesReg(AsBase + (p ^ 1) * ABUF, BsBase + (p ^ 1) * BBUF);
      waitAsync();
    }
    __syncthreads();
    p ^= 1;
  }

  // ---- epilogue: D row = r + 8*half, col = lane%16; ACT is compile-time ----
#pragma unroll
  for (int c = 0; c < 2; ++c) {
    int col = bn + wc * 32 + c * 16 + lrow;
    if (!GUARD || col < N) {
      float bv = bp ? bp[col] : 0.f;
#pragma unroll
      for (int r = 0; r < 8; ++r) {
        int row = bm + wr * 16 + r + 8 * half;
        if (!GUARD || row < M) {
          float v = acc[c][r] + bv;
          if (ACT == 1) v = fmaxf(v, 0.f);
          if (ACT == 2) v = 0.5f * v * (1.f + erff(v * 0.70710678118654752f));
          C[(long)row * N + col] = (OT)v;
        }
      }
    }
  }
}

template <int ACT, typename AT, typename OT>
__global__ __launch_bounds__(GT) void gemm_wmma(
    const AT* __restrict__ A, const float* __restrict__ B,
    const float* __restrict__ bias, OT* __restrict__ C,
    int M, int N, int K,
    long sA, long sB, long sBias, long sC)
{
  __shared__ __attribute__((aligned(16))) __bf16 As[2 * ABUF];
  __shared__ __attribute__((aligned(16))) __bf16 Bs[2 * BBUF];

  const int z = blockIdx.z;
  A += (long)z * sA;
  B += (long)z * sB;
  C += (long)z * sC;
  const float* bp = bias ? (bias + (long)z * sBias) : nullptr;
  const int bm = blockIdx.y * TMv;
  const int bn = blockIdx.x * TNv;

  if (bm + TMv <= M && bn + TNv <= N)
    gemm_body<false, ACT, AT, OT>(A, B, bp, C, M, N, K, As, Bs, bm, bn);
  else
    gemm_body<true, ACT, AT, OT>(A, B, bp, C, M, N, K, As, Bs, bm, bn);
}

// ---------------- elementwise / graph / routing kernels ----------------
__global__ void k_fill_f(float* p, float v, long n) {
  long i = (long)blockIdx.x * blockDim.x + threadIdx.x;
  if (i < n) p[i] = v;
}
__global__ void k_zero_bf(__bf16* p, long n) {
  long i = (long)blockIdx.x * blockDim.x + threadIdx.x;
  if (i < n) p[i] = (__bf16)0.f;
}
__global__ void k_zero_i(int* p, long n) {
  long i = (long)blockIdx.x * blockDim.x + threadIdx.x;
  if (i < n) p[i] = 0;
}
__global__ void k_copy_f(float* d, const float* s, long n) {
  long i = (long)blockIdx.x * blockDim.x + threadIdx.x;
  if (i < n) d[i] = s[i];
}
__global__ void k_deg(const int* ei, float* deg, int E) {
  int e = blockIdx.x * blockDim.x + threadIdx.x;
  if (e < E) atomicAdd(&deg[ei[E + e]], 1.f);   // edge_index[1][e]
}
__global__ void k_rsqrt(float* d, int n) {
  int i = blockIdx.x * blockDim.x + threadIdx.x;
  if (i < n) d[i] = rsqrtf(d[i]);
}
// self-loop contribution: agg = hW * dinv^2  (plain store, runs first)
__global__ void k_selfloop(const float* hW, const float* dinv, float* agg,
                           int n, int d) {
  long idx = (long)blockIdx.x * blockDim.x + threadIdx.x;
  if (idx >= (long)n * d) return;
  int i = (int)(idx / d);
  float w = dinv[i] * dinv[i];
  agg[idx] = hW[idx] * w;
}
__global__ void k_edge_msg(const int* ei, const float* hW, const float* dinv,
                           float* agg, int E, int d) {
  long idx = (long)blockIdx.x * blockDim.x + threadIdx.x;
  if (idx >= (long)E * d) return;
  int e = (int)(idx / d), f = (int)(idx % d);
  int s = ei[e], dst = ei[E + e];
  float w = dinv[s] * dinv[dst];
  atomicAdd(&agg[(long)dst * d + f], hW[(long)s * d + f] * w);
}
// h = relu(agg + bc + (x@Wr + br))
__global__ void k_combine_relu(const float* agg, const float* bc,
                               const float* rW, float* h, int n, int d) {
  long idx = (long)blockIdx.x * blockDim.x + threadIdx.x;
  if (idx >= (long)n * d) return;
  int f = (int)(idx % d);
  float v = agg[idx] + bc[f] + rW[idx];
  h[idx] = fmaxf(v, 0.f);
}
__global__ void k_top2(const float* logits, int* e0, int* e1,
                       float* g0, float* g1, int n, int ne) {
  int i = blockIdx.x * blockDim.x + threadIdx.x;
  if (i >= n) return;
  const float* row = logits + (long)i * ne;
  float v0 = -3.4e38f, v1 = -3.4e38f;
  int i0 = 0, i1 = 0;
  for (int j = 0; j < ne; ++j) {
    float v = row[j];
    if (v > v0) { v1 = v0; i1 = i0; v0 = v; i0 = j; }
    else if (v > v1) { v1 = v; i1 = j; }
  }
  float e = __expf(v1 - v0);
  float s = 1.f + e;
  e0[i] = i0; e1[i] = i1;
  g0[i] = 1.f / s; g1[i] = e / s;
}
__global__ void k_route(const int* e0, const int* e1, int* cnt, int* posArr,
                        int n) {
  int t = blockIdx.x * blockDim.x + threadIdx.x;
  if (t >= 2 * n) return;
  int node = t >> 1, k = t & 1;
  int e = k ? e1[node] : e0[node];
  posArr[t] = atomicAdd(&cnt[e], 1);
}
__global__ void k_gather(const float* h, const int* e0, const int* e1,
                         const int* posArr, __bf16* buf, int n, int d) {
  long idx = (long)blockIdx.x * blockDim.x + threadIdx.x;
  if (idx >= (long)2 * n * d) return;
  int t = (int)(idx / d), f = (int)(idx % d);
  int node = t >> 1, k = t & 1;
  int e = k ? e1[node] : e0[node];
  int pos = posArr[t];
  if (pos < CAP)
    buf[((long)e * CAP + pos) * d + f] = (__bf16)h[(long)node * d + f];
}
__global__ void k_scatter(const __bf16* y, const int* e0, const int* e1,
                          const float* g0, const float* g1, const int* posArr,
                          float* pre, int n, int d) {
  long idx = (long)blockIdx.x * blockDim.x + threadIdx.x;
  if (idx >= (long)2 * n * d) return;
  int t = (int)(idx / d), f = (int)(idx % d);
  int node = t >> 1, k = t & 1;
  int e = k ? e1[node] : e0[node];
  int pos = posArr[t];
  float g = k ? g1[node] : g0[node];
  if (pos < CAP)
    atomicAdd(&pre[(long)node * d + f],
              g * (float)y[((long)e * CAP + pos) * d + f]);
}
__global__ void k_bn_stats(const float* pre, float* mean, float* rstd,
                           int n, int d) {
  __shared__ float ss[256], ss2[256];
  int f = blockIdx.x;
  float s = 0.f, s2 = 0.f;
  for (int i = threadIdx.x; i < n; i += blockDim.x) {
    float v = pre[(long)i * d + f];
    s += v; s2 += v * v;
  }
  ss[threadIdx.x] = s; ss2[threadIdx.x] = s2;
  __syncthreads();
  for (int st = 128; st > 0; st >>= 1) {
    if (threadIdx.x < st) {
      ss[threadIdx.x]  += ss[threadIdx.x + st];
      ss2[threadIdx.x] += ss2[threadIdx.x + st];
    }
    __syncthreads();
  }
  if (threadIdx.x == 0) {
    float mu  = ss[0] / (float)n;
    float var = ss2[0] / (float)n - mu * mu;
    mean[f] = mu;
    rstd[f] = rsqrtf(var + BN_EPS);
  }
}
__global__ void k_bn_apply(const float* pre, const float* mean,
                           const float* rstd, const float* gamma,
                           const float* beta, float* out, int n, int d) {
  long idx = (long)blockIdx.x * blockDim.x + threadIdx.x;
  if (idx >= (long)n * d) return;
  int f = (int)(idx % d);
  out[idx] = (pre[idx] - mean[f]) * rstd[f] * gamma[f] + beta[f];
}

// ---------------- host-side orchestration ----------------
static inline dim3 gemm_grid(int M, int N, int batch) {
  return dim3((N + TNv - 1) / TNv, (M + TMv - 1) / TMv, batch);
}
template <int ACT>
static void gemm_f32(hipStream_t st, const float* A, const float* B,
                     const float* bias, float* C, int M, int N, int K) {
  gemm_wmma<ACT, float, float><<<gemm_grid(M, N, 1), GT, 0, st>>>(
      A, B, bias, C, M, N, K, 0, 0, 0, 0);
}
template <int ACT>
static void gemm_bf(hipStream_t st, const __bf16* A, const float* B,
                    const float* bias, __bf16* C, int M, int N, int K,
                    int batch, long sA, long sB, long sBias, long sC) {
  gemm_wmma<ACT, __bf16, __bf16><<<gemm_grid(M, N, batch), GT, 0, st>>>(
      A, B, bias, C, M, N, K, sA, sB, sBias, sC);
}

struct Scratch {
  float *dinv, *hW, *rW, *agg, *h, *logits, *g0, *g1, *pre, *mean, *rstd;
  int *e0, *e1, *posArr, *cnt;
  __bf16 *buf, *hid, *ybuf;
};

#define LAUNCH_EW(kern, count, ...)                                          \
  kern<<<(int)(((count) + 255) / 256), 256, 0, st>>>(__VA_ARGS__)

static void run_block(hipStream_t st, Scratch& S, const float* xin, int inF,
                      const int* ei, const float* Wc, const float* bc,
                      const float* Wr, const float* br, const float* Wg,
                      const float* bg, const float* W1, const float* b1,
                      const float* W2, const float* b2, const float* gamma,
                      const float* beta, float* xout) {
  const int n = N_NODES, d = HID;
  const long nd = (long)n * d;

  // GCN conv
  gemm_f32<0>(st, xin, Wc, nullptr, S.hW, n, d, inF);
  gemm_f32<0>(st, xin, Wr, br, S.rW, n, d, inF);
  LAUNCH_EW(k_selfloop, nd, S.hW, S.dinv, S.agg, n, d);
  LAUNCH_EW(k_edge_msg, (long)N_EDGES * d, ei, S.hW, S.dinv, S.agg, N_EDGES, d);
  LAUNCH_EW(k_combine_relu, nd, S.agg, bc, S.rW, S.h, n, d);

  // MoE gate + routing
  gemm_f32<0>(st, S.h, Wg, bg, S.logits, n, N_EXP, d);
  LAUNCH_EW(k_top2, n, S.logits, S.e0, S.e1, S.g0, S.g1, n, N_EXP);
  LAUNCH_EW(k_zero_i, N_EXP, S.cnt, (long)N_EXP);
  LAUNCH_EW(k_route, 2 * n, S.e0, S.e1, S.cnt, S.posArr, n);

  // expert buffers (bf16) + batched expert GEMMs
  const long bufN = (long)N_EXP * CAP * d;
  LAUNCH_EW(k_zero_bf, bufN, S.buf, bufN);
  LAUNCH_EW(k_gather, (long)2 * n * d, S.h, S.e0, S.e1, S.posArr, S.buf, n, d);
  gemm_bf<2>(st, S.buf, W1, b1, S.hid, CAP, 2 * d, d, N_EXP,
             (long)CAP * d, (long)d * 2 * d, 2 * d, (long)CAP * 2 * d);
  gemm_bf<0>(st, S.hid, W2, b2, S.ybuf, CAP, d, 2 * d, N_EXP,
             (long)CAP * 2 * d, (long)2 * d * d, d, (long)CAP * d);

  // combine + batchnorm
  LAUNCH_EW(k_copy_f, nd, S.pre, S.h, nd);
  LAUNCH_EW(k_scatter, (long)2 * n * d, S.ybuf, S.e0, S.e1, S.g0, S.g1,
            S.posArr, S.pre, n, d);
  k_bn_stats<<<d, 256, 0, st>>>(S.pre, S.mean, S.rstd, n, d);
  LAUNCH_EW(k_bn_apply, nd, S.pre, S.mean, S.rstd, gamma, beta, xout, n, d);
}

extern "C" void kernel_launch(void* const* d_in, const int* in_sizes, int n_in,
                              void* d_out, int out_size, void* d_ws,
                              size_t ws_size, hipStream_t stream) {
  (void)in_sizes; (void)n_in; (void)out_size; (void)ws_size;
  hipStream_t st = stream;
  const float* x  = (const float*)d_in[0];
  const int*   ei = (const int*)d_in[1];
  auto F = [&](int i) { return (const float*)d_in[i]; };

  // carve workspace
  char* base = (char*)d_ws;
  size_t off = 0;
  auto carve = [&](size_t bytes) -> char* {
    char* r = base + off;
    off = (off + bytes + 255) & ~(size_t)255;
    return r;
  };
  Scratch S;
  S.dinv   = (float*)carve((size_t)N_NODES * 4);
  S.hW     = (float*)carve((size_t)N_NODES * HID * 4);
  S.rW     = (float*)carve((size_t)N_NODES * HID * 4);
  S.agg    = (float*)carve((size_t)N_NODES * HID * 4);
  S.h      = (float*)carve((size_t)N_NODES * HID * 4);
  S.logits = (float*)carve((size_t)N_NODES * N_EXP * 4);
  S.g0     = (float*)carve((size_t)N_NODES * 4);
  S.g1     = (float*)carve((size_t)N_NODES * 4);
  S.pre    = (float*)carve((size_t)N_NODES * HID * 4);
  S.mean   = (float*)carve((size_t)HID * 4);
  S.rstd   = (float*)carve((size_t)HID * 4);
  S.e0     = (int*)carve((size_t)N_NODES * 4);
  S.e1     = (int*)carve((size_t)N_NODES * 4);
  S.posArr = (int*)carve((size_t)N_NODES * 2 * 4);
  S.cnt    = (int*)carve((size_t)N_EXP * 4);
  S.buf    = (__bf16*)carve((size_t)N_EXP * CAP * HID * 2);
  S.hid    = (__bf16*)carve((size_t)N_EXP * CAP * 2 * HID * 2);
  S.ybuf   = (__bf16*)carve((size_t)N_EXP * CAP * HID * 2);
  float* x1 = (float*)carve((size_t)N_NODES * HID * 4);
  float* x2 = (float*)carve((size_t)N_NODES * HID * 4);

  // symmetric GCN normalization: deg includes self-loop
  k_fill_f<<<(N_NODES + 255) / 256, 256, 0, st>>>(S.dinv, 1.f, N_NODES);
  k_deg<<<(N_EDGES + 255) / 256, 256, 0, st>>>(ei, S.dinv, N_EDGES);
  k_rsqrt<<<(N_NODES + 255) / 256, 256, 0, st>>>(S.dinv, N_NODES);

  // block 0 (in 512 -> 256), block 1 (256 -> 256)
  run_block(st, S, x, IN_F, ei, F(2), F(3), F(4), F(5), F(6), F(7), F(8),
            F(9), F(10), F(11), F(12), F(13), x1);
  run_block(st, S, x1, HID, ei, F(14), F(15), F(16), F(17), F(18), F(19),
            F(20), F(21), F(22), F(23), F(24), F(25), x2);

  // final classifier
  gemm_f32<0>(st, x2, F(26), F(27), (float*)d_out, N_NODES, OUT_F, HID);
}
